// GATV2_79328045957509
// MI455X (gfx1250) — compile-verified, matrix operand order checked by reading, not run on hardware
//
#include <hip/hip_runtime.h>

// ---------------------------------------------------------------------------
// GATv2 (B=8, N=1024, F=128, NHID1=128, 8 heads x 16; layer2: 1 head x 256)
// e[i,j] = s_l[i] + s_r[j]  =>  after masked softmax s_l cancels, so
//   hp = (A @ [w (.) WhR | w]) row-normalized,  w = exp(s_r - max s_r).
// All torch reshapes are contiguous flat views.  Heavy GEMMs run on
// v_wmma_f32_16x16x32_f16 with 2x2 register blocking (32x32 per wave32);
// the attention GEMM stages its 64KB B-strip into LDS via the Tensor Data
// Mover (tensor_load_to_lds + s_wait_tensorcnt) and reads it with ds_load.
// ---------------------------------------------------------------------------

typedef _Float16     v8h  __attribute__((ext_vector_type(8)));
typedef _Float16     v16h __attribute__((ext_vector_type(16)));
typedef float        v8f  __attribute__((ext_vector_type(8)));
typedef unsigned int v4u  __attribute__((ext_vector_type(4)));
typedef int          v4i  __attribute__((ext_vector_type(4)));
typedef int          v8i  __attribute__((ext_vector_type(8)));

#define SLOPE 0.2f

#if defined(__has_builtin)
#if __has_builtin(__builtin_amdgcn_tensor_load_to_lds)
#define HAVE_TDM 1
#else
#define HAVE_TDM 0
#endif
#else
#define HAVE_TDM 0
#endif

__device__ __forceinline__ float elu_f(float x)   { return x > 0.f ? x : (__expf(x) - 1.f); }
__device__ __forceinline__ float leaky_f(float x) { return x > 0.f ? x : SLOPE * x; }

__device__ __forceinline__ v16h cat16(v8h lo, v8h hi) {
    return __builtin_shufflevector(lo, hi, 0,1,2,3,4,5,6,7,8,9,10,11,12,13,14,15);
}

#if HAVE_TDM
// TDM: 2D tile load global -> LDS. rows x rowlen f16 elements, row stride in
// elements.  D# packed per CDNA5 ISA 8.3/8.4 (Group0: count=1, type=2;
// Group1: data_size=1 (2B), tensor_dim0/1, tile_dim0/1, dim0 stride).
__device__ __forceinline__ void tdm_load_2d_f16(unsigned lds_addr, const void* gaddr,
                                                unsigned rows, unsigned rowlen,
                                                unsigned stride_elems) {
    unsigned long long ga = (unsigned long long)gaddr;
    v4u g0 = {};
    g0[0] = 1u;                                           // count=1
    g0[1] = lds_addr;                                     // lds_addr (bytes)
    g0[2] = (unsigned)(ga & 0xFFFFFFFFu);                 // global_addr[31:0]
    g0[3] = (unsigned)((ga >> 32) & 0x1FFFFFFu) | (2u << 30); // [56:32] | type=2
    v8i g1 = {};
    g1[0] = (int)(1u << 16);                              // data_size=1 => 2 bytes
    g1[1] = (int)((rowlen & 0xFFFFu) << 16);              // tensor_dim0[15:0]
    g1[2] = (int)(((rowlen >> 16) & 0xFFFFu) | ((rows & 0xFFFFu) << 16)); // dim0 hi | dim1 lo
    g1[3] = (int)(((rows >> 16) & 0xFFFFu) | ((rowlen & 0xFFFFu) << 16)); // dim1 hi | tile_dim0
    g1[4] = (int)(rows & 0xFFFFu);                        // tile_dim1 (tile_dim2=0)
    g1[5] = (int)stride_elems;                            // tensor_dim0_stride[31:0]
    g1[6] = 0;
    g1[7] = 0;
    v4i gz = {};
#if __clang_major__ >= 23
    v8i gz8 = {};
    __builtin_amdgcn_tensor_load_to_lds(g0, g1, gz, gz, gz8, 0);
#else
    __builtin_amdgcn_tensor_load_to_lds(g0, g1, gz, gz, 0);
#endif
}
#endif

// ---------------------------------------------------------------------------
// graph_norm: x viewed as [B, C=128, 1024] (pure reshape => contiguous 1024
// chunks).  One block per (b,c) chunk.  Output converted to f16.
// ---------------------------------------------------------------------------
__global__ void gnorm_kernel(const float* __restrict__ x,
                             const float* __restrict__ scale,
                             const float* __restrict__ shift,
                             const float* __restrict__ alpha,
                             _Float16* __restrict__ out) {
    __shared__ float sA[256];
    __shared__ float sB[256];
    const int blk = blockIdx.x;          // b*128 + c
    const int c   = blk & 127;
    const int tid = threadIdx.x;
    const float* xp = x + (size_t)blk * 1024;

    float sum = 0.f, sq = 0.f;
    for (int i = tid; i < 1024; i += 256) {
        float v = xp[i];
        sum += v; sq += v * v;
    }
    sA[tid] = sum; sB[tid] = sq;
    __syncthreads();
    for (int off = 128; off > 0; off >>= 1) {
        if (tid < off) { sA[tid] += sA[tid + off]; sB[tid] += sB[tid + off]; }
        __syncthreads();
    }
    const float mean = sA[0] * (1.f / 1024.f);
    const float var  = sB[0] * (1.f / 1024.f) - mean * mean;
    const float rs   = rsqrtf(var + 1e-5f);
    const float al = alpha[c], sc = scale[c], sh = shift[c];
    _Float16* op = out + (size_t)blk * 1024;
    for (int i = tid; i < 1024; i += 256) {
        float v = (xp[i] - al * mean) * rs;
        op[i] = (_Float16)(sc * v + sh);
    }
}

__global__ void adj_kernel(const int* __restrict__ adj, _Float16* __restrict__ out, int n) {
    int i = blockIdx.x * blockDim.x + threadIdx.x;
    if (i < n) out[i] = (_Float16)(adj[i] > 0 ? 1.0f : 0.0f);
}

__global__ void cvt_transpose_kernel(const float* __restrict__ in, _Float16* __restrict__ out,
                                     int rows, int cols) {
    int i = blockIdx.x * blockDim.x + threadIdx.x;
    if (i < rows * cols) {
        int r = i / cols, c = i % cols;
        out[(size_t)c * rows + r] = (_Float16)in[i];
    }
}

__global__ void cvt_vec_kernel(const float* __restrict__ in, _Float16* __restrict__ out, int n) {
    int i = blockIdx.x * blockDim.x + threadIdx.x;
    if (i < n) out[i] = (_Float16)in[i];
}

// ---------------------------------------------------------------------------
// Feature GEMM: out = leaky_relu(A[M,128] @ B[128,Nc]) in f16, BT[Nc,128].
// 32x32 output per wave (2x2 WMMA tiles), fragments straight from global/L2.
// ---------------------------------------------------------------------------
__global__ void gemm_feat_kernel(const _Float16* __restrict__ A,
                                 const _Float16* __restrict__ BT,
                                 _Float16* __restrict__ out,
                                 int M, int Nc) {
    const int K = 128;
    const int lane  = threadIdx.x & 31;
    const int wave  = threadIdx.x >> 5;
    const int tN2   = Nc >> 5;                    // 32-col blocks
    const int tM2   = M >> 5;
    const int tile  = blockIdx.x * (blockDim.x >> 5) + wave;
    if (tile >= tM2 * tN2) return;
    const int mt2 = tile / tN2, nt2 = tile % tN2;

    const int idx15 = lane & 15;
    const int khalf = (lane >> 4) * 8;
    const _Float16* A0 = A  + (size_t)(mt2 * 32 + idx15) * K;
    const _Float16* A1 = A0 + (size_t)16 * K;
    const _Float16* B0 = BT + (size_t)(nt2 * 32 + idx15) * K;
    const _Float16* B1 = B0 + (size_t)16 * K;

    v8f acc00 = {}, acc01 = {}, acc10 = {}, acc11 = {};
    #pragma unroll
    for (int k0 = 0; k0 < K; k0 += 32) {
        v16h a0 = cat16(*(const v8h*)(A0 + k0 + khalf), *(const v8h*)(A0 + k0 + 16 + khalf));
        v16h a1 = cat16(*(const v8h*)(A1 + k0 + khalf), *(const v8h*)(A1 + k0 + 16 + khalf));
        v16h b0 = cat16(*(const v8h*)(B0 + k0 + khalf), *(const v8h*)(B0 + k0 + 16 + khalf));
        v16h b1 = cat16(*(const v8h*)(B1 + k0 + khalf), *(const v8h*)(B1 + k0 + 16 + khalf));
        acc00 = __builtin_amdgcn_wmma_f32_16x16x32_f16(false, a0, false, b0, (short)0, acc00, false, false);
        acc01 = __builtin_amdgcn_wmma_f32_16x16x32_f16(false, a0, false, b1, (short)0, acc01, false, false);
        acc10 = __builtin_amdgcn_wmma_f32_16x16x32_f16(false, a1, false, b0, (short)0, acc10, false, false);
        acc11 = __builtin_amdgcn_wmma_f32_16x16x32_f16(false, a1, false, b1, (short)0, acc11, false, false);
    }
    const int mb = (lane >> 4) * 8;
    #pragma unroll
    for (int r = 0; r < 8; ++r) {
        size_t r0 = (size_t)(mt2 * 32 + mb + r) * Nc;
        size_t r1 = (size_t)(mt2 * 32 + 16 + mb + r) * Nc;
        int    c0 = nt2 * 32 + idx15;
        out[r0 + c0]      = (_Float16)leaky_f(acc00[r]);
        out[r0 + c0 + 16] = (_Float16)leaky_f(acc01[r]);
        out[r1 + c0]      = (_Float16)leaky_f(acc10[r]);
        out[r1 + c0 + 16] = (_Float16)leaky_f(acc11[r]);
    }
}

// ---------------------------------------------------------------------------
// Attention aggregation GEMM: per bh: C[1024,Nc] = adjf[b] @ B(bh), with B
// supplied transposed bt[bh][Nc][1024].  Block = 256 rows x 32-col strip;
// the strip (32x1024 f16 = 64KB) is staged to LDS by the TDM once and reused
// by all 8 waves; each wave computes a 32x32 tile with 4 WMMA accumulators.
// Nc must be a multiple of 32.
// ---------------------------------------------------------------------------
__global__ void __launch_bounds__(256)
gemm_attn_kernel(const _Float16* __restrict__ adjf, // [B][1024][1024]
                 const _Float16* __restrict__ bt,   // [BH][Nc][1024]
                 float* __restrict__ c,             // [BH][1024][Nc]
                 int H, int Nc) {
    const int K = 1024, M = 1024;
    __shared__ __align__(16) _Float16 Bs[32 * 1024];   // 64KB strip

    const int lane = threadIdx.x & 31;
    const int wave = threadIdx.x >> 5;
    const int nstrips = Nc >> 5;
    const int mblocks = M >> 8;                        // 256-row blocks = 4
    const int blk = blockIdx.x;
    const int bh  = blk / (mblocks * nstrips);
    const int r0  = blk % (mblocks * nstrips);
    const int mb  = r0 / nstrips;                      // 256-row block index
    const int ns  = r0 % nstrips;                      // 32-col strip index
    const int b   = bh / H;

    const _Float16* strip = bt + ((size_t)bh * Nc + (size_t)ns * 32) * K;
#if HAVE_TDM
    if (wave == 0) {
        tdm_load_2d_f16((unsigned)(unsigned long long)(void*)Bs, strip, 32u, 1024u, 1024u);
        __builtin_amdgcn_s_wait_tensorcnt((short)0);
    }
#else
    for (int i = threadIdx.x; i < 32 * 1024 / 8; i += 256)
        ((v8h*)Bs)[i] = ((const v8h*)strip)[i];
#endif
    __syncthreads();

    const int idx15 = lane & 15;
    const int khalf = (lane >> 4) * 8;
    const _Float16* A0 = adjf + (size_t)b * M * K + (size_t)(mb * 256 + wave * 32 + idx15) * K;
    const _Float16* A1 = A0 + (size_t)16 * K;
    const _Float16* L0 = Bs + (size_t)idx15 * K;        // strip cols 0..15
    const _Float16* L1 = Bs + (size_t)(16 + idx15) * K; // strip cols 16..31

    v8f acc00 = {}, acc01 = {}, acc10 = {}, acc11 = {};
    for (int k0 = 0; k0 < K; k0 += 32) {
        if (k0 + 128 < K) {
            __builtin_prefetch(A0 + k0 + 128, 0, 1);
            __builtin_prefetch(A1 + k0 + 128, 0, 1);
        }
        v16h a0 = cat16(*(const v8h*)(A0 + k0 + khalf), *(const v8h*)(A0 + k0 + 16 + khalf));
        v16h a1 = cat16(*(const v8h*)(A1 + k0 + khalf), *(const v8h*)(A1 + k0 + 16 + khalf));
        v16h b0 = cat16(*(const v8h*)(L0 + k0 + khalf), *(const v8h*)(L0 + k0 + 16 + khalf));
        v16h b1 = cat16(*(const v8h*)(L1 + k0 + khalf), *(const v8h*)(L1 + k0 + 16 + khalf));
        acc00 = __builtin_amdgcn_wmma_f32_16x16x32_f16(false, a0, false, b0, (short)0, acc00, false, false);
        acc01 = __builtin_amdgcn_wmma_f32_16x16x32_f16(false, a0, false, b1, (short)0, acc01, false, false);
        acc10 = __builtin_amdgcn_wmma_f32_16x16x32_f16(false, a1, false, b0, (short)0, acc10, false, false);
        acc11 = __builtin_amdgcn_wmma_f32_16x16x32_f16(false, a1, false, b1, (short)0, acc11, false, false);
    }

    float* Cbh = c + (size_t)bh * M * Nc;
    const int mbase = (lane >> 4) * 8;
    const int row0  = mb * 256 + wave * 32;
    const int col0  = ns * 32 + idx15;
    #pragma unroll
    for (int r = 0; r < 8; ++r) {
        size_t ra = (size_t)(row0 + mbase + r) * Nc;
        size_t rb = (size_t)(row0 + 16 + mbase + r) * Nc;
        Cbh[ra + col0]      = acc00[r];
        Cbh[ra + col0 + 16] = acc01[r];
        Cbh[rb + col0]      = acc10[r];
        Cbh[rb + col0 + 16] = acc11[r];
    }
}

// per-row score: s[r] = dot(Wh[r, 0:D], a)
__global__ void score_kernel(const _Float16* __restrict__ wh, const _Float16* __restrict__ a,
                             float* __restrict__ s, int rows, int D) {
    int r = blockIdx.x * blockDim.x + threadIdx.x;
    if (r < rows) {
        const _Float16* p = wh + (size_t)r * D;
        float acc = 0.f;
        for (int d = 0; d < D; ++d) acc += (float)p[d] * (float)a[d];
        s[r] = acc;
    }
}

// per group of 1024: w = exp(s - max(s))
__global__ void maxexp_kernel(const float* __restrict__ s, float* __restrict__ w) {
    __shared__ float red[256];
    const int tid = threadIdx.x;
    const float* sp = s + (size_t)blockIdx.x * 1024;
    float m = -3.0e38f;
    for (int i = tid; i < 1024; i += 256) m = fmaxf(m, sp[i]);
    red[tid] = m;
    __syncthreads();
    for (int off = 128; off > 0; off >>= 1) {
        if (tid < off) red[tid] = fmaxf(red[tid], red[tid + off]);
        __syncthreads();
    }
    const float mx = red[0];
    float* wp = w + (size_t)blockIdx.x * 1024;
    for (int i = tid; i < 1024; i += 256) wp[i] = __expf(sp[i] - mx);
}

// bt[bh][col][j]: col<D: w*wh; col==D: w (denominator); col>D: 0 (pad)
__global__ void build_bt_kernel(const _Float16* __restrict__ wh, const float* __restrict__ w,
                                _Float16* __restrict__ bt, int BH, int D, int Ncols) {
    long i = (long)blockIdx.x * blockDim.x + threadIdx.x;
    long total = (long)BH * Ncols * 1024;
    if (i >= total) return;
    int j   = (int)(i & 1023);
    long t  = i >> 10;
    int col = (int)(t % Ncols);
    int bh  = (int)(t / Ncols);
    float wv = w[(size_t)bh * 1024 + j];
    float val;
    if (col < D)       val = wv * (float)wh[((size_t)bh * 1024 + j) * D + col];
    else if (col == D) val = wv;
    else               val = 0.f;
    bt[i] = (_Float16)val;
}

// layer-1 epilogue: h2pre[i] = elu(C1[bh][node][d] / C1[bh][node][16]),
// flat i = bh*16384 + node*16 + d (== torch reshape-back layout)
__global__ void epi1_kernel(const float* __restrict__ c1, float* __restrict__ h2pre) {
    long i = (long)blockIdx.x * blockDim.x + threadIdx.x;  // < 8*8*1024*16
    int bh   = (int)(i >> 14);
    int rem  = (int)(i & 16383);
    int node = rem >> 4;
    int d    = rem & 15;
    const float* row = c1 + (size_t)bh * 1024 * 32 + (size_t)node * 32;
    h2pre[i] = elu_f(row[d] / row[16]);
}

// layer-2 epilogue (C2 row stride 288, denominator col 256)
__global__ void epi2_kernel(const float* __restrict__ c2, float* __restrict__ out) {
    long i = (long)blockIdx.x * blockDim.x + threadIdx.x;  // < 8*1024*256
    int b    = (int)(i >> 18);
    int rem  = (int)(i & 262143);
    int node = rem >> 8;
    int d    = rem & 255;
    const float* row = c2 + (size_t)b * 1024 * 288 + (size_t)node * 288;
    out[i] = elu_f(row[d] / row[256]);
}

// ---------------------------------------------------------------------------
static inline size_t align_up(size_t x) { return (x + 255) & ~(size_t)255; }

extern "C" void kernel_launch(void* const* d_in, const int* in_sizes, int n_in,
                              void* d_out, int out_size, void* d_ws, size_t ws_size,
                              hipStream_t stream) {
    const float* x     = (const float*)d_in[0];
    const int*   adj   = (const int*)  d_in[1];
    // d_in[2] = W_l1 (unused: s_l cancels in softmax)
    const float* W_r1  = (const float*)d_in[3];
    const float* a1    = (const float*)d_in[4];
    // d_in[5] = W_l2 (unused)
    const float* W_r2  = (const float*)d_in[6];
    const float* a2    = (const float*)d_in[7];
    const float* gn1_scale = (const float*)d_in[8];
    const float* gn1_shift = (const float*)d_in[9];
    const float* gn1_alpha = (const float*)d_in[10];
    const float* gn2_scale = (const float*)d_in[11];
    const float* gn2_shift = (const float*)d_in[12];
    const float* gn2_alpha = (const float*)d_in[13];

    char* ws = (char*)d_ws;
    size_t off = 0;
    auto carve = [&](size_t bytes) { char* p = ws + off; off = align_up(off + bytes); return p; };

    _Float16* h1f16 = (_Float16*)carve((size_t)2 * 8192 * 128);
    _Float16* adjf  = (_Float16*)carve((size_t)2 * 8 * 1024 * 1024);
    _Float16* wrt1  = (_Float16*)carve((size_t)2 * 128 * 128);
    _Float16* wrt2  = (_Float16*)carve((size_t)2 * 256 * 128);
    _Float16* a1h   = (_Float16*)carve((size_t)2 * 16);
    _Float16* a2h   = (_Float16*)carve((size_t)2 * 256);
    _Float16* whr1  = (_Float16*)carve((size_t)2 * 8192 * 128);
    float*    s1    = (float*)   carve((size_t)4 * 65536);
    float*    w1    = (float*)   carve((size_t)4 * 65536);
    _Float16* b1t   = (_Float16*)carve((size_t)2 * 64 * 32 * 1024);
    float*    c1    = (float*)   carve((size_t)4 * 64 * 1024 * 32);
    float*    h2pre = (float*)   carve((size_t)4 * 8192 * 128);
    _Float16* h2n   = (_Float16*)carve((size_t)2 * 8192 * 128);
    _Float16* whr2  = (_Float16*)carve((size_t)2 * 8192 * 256);
    float*    s2    = (float*)   carve((size_t)4 * 8192);
    float*    w2    = (float*)   carve((size_t)4 * 8192);
    _Float16* b2t   = (_Float16*)carve((size_t)2 * 8 * 288 * 1024);   // Nc padded to 288
    float*    c2    = (float*)   carve((size_t)4 * 8 * 1024 * 288);
    (void)ws_size; (void)in_sizes; (void)n_in; (void)out_size;

    // ---- stage 0 -------------------------------------------------------------
    gnorm_kernel<<<1024, 256, 0, stream>>>(x, gn1_scale, gn1_shift, gn1_alpha, h1f16);
    adj_kernel<<<(8 * 1024 * 1024) / 256, 256, 0, stream>>>(adj, adjf, 8 * 1024 * 1024);
    cvt_transpose_kernel<<<64, 256, 0, stream>>>(W_r1, wrt1, 128, 128);
    cvt_transpose_kernel<<<128, 256, 0, stream>>>(W_r2, wrt2, 128, 256);
    cvt_vec_kernel<<<1, 32, 0, stream>>>(a1, a1h, 16);
    cvt_vec_kernel<<<1, 256, 0, stream>>>(a2, a2h, 256);

    // ---- layer 1 -------------------------------------------------------------
    gemm_feat_kernel<<<128, 256, 0, stream>>>(h1f16, wrt1, whr1, 8192, 128);     // 1024 wave tiles
    score_kernel<<<256, 256, 0, stream>>>(whr1, a1h, s1, 65536, 16);
    maxexp_kernel<<<64, 256, 0, stream>>>(s1, w1);
    build_bt_kernel<<<8192, 256, 0, stream>>>(whr1, w1, b1t, 64, 16, 32);
    gemm_attn_kernel<<<256, 256, 0, stream>>>(adjf, b1t, c1, 8, 32);             // 64 bh x 4 mblk
    epi1_kernel<<<4096, 256, 0, stream>>>(c1, h2pre);

    // ---- layer 2 -------------------------------------------------------------
    gnorm_kernel<<<1024, 256, 0, stream>>>(h2pre, gn2_scale, gn2_shift, gn2_alpha, h2n);
    gemm_feat_kernel<<<256, 256, 0, stream>>>(h2n, wrt2, whr2, 8192, 256);
    score_kernel<<<32, 256, 0, stream>>>(whr2, a2h, s2, 8192, 256);
    maxexp_kernel<<<8, 256, 0, stream>>>(s2, w2);
    build_bt_kernel<<<9216, 256, 0, stream>>>(whr2, w2, b2t, 8, 256, 288);
    gemm_attn_kernel<<<288, 256, 0, stream>>>(adjf, b2t, c2, 1, 288);            // 8 b x 4 x 9
    epi2_kernel<<<8192, 256, 0, stream>>>(c2, (float*)d_out);
}
